// NeuralODE_8787503087900
// MI455X (gfx1250) — matrix operand order repaired and checked by reading, not compiled, not run on hardware
//
#include <hip/hip_runtime.h>
#include <hip/hip_bf16.h>

// ---------------------------------------------------------------------------
// Problem constants (match reference)
// ---------------------------------------------------------------------------
#define BDIM 2048
#define DDIM 512
#define SDIM 64
#define HDIM 2048
#define TDIM 16
#define STEPS_PER_INTERVAL 4

typedef __attribute__((ext_vector_type(16))) _Float16 v16h;
typedef __attribute__((ext_vector_type(8)))  float    v8f;
typedef __attribute__((ext_vector_type(8)))  _Float16 half8;
typedef __attribute__((ext_vector_type(4)))  int      v4i;

union Frag16 { half8 h[2]; v16h v; };

// LDS tile row stride in halves: 32 payload + 8 pad -> 80B rows (16B aligned,
// spreads the 16-row strided b128 fragment reads across banks)
#define LDS_STRIDE 40

#if __has_builtin(__builtin_amdgcn_global_load_async_to_lds_b128)
#define HAVE_ASYNC_LDS 1
#else
#define HAVE_ASYNC_LDS 0
#endif

typedef __attribute__((address_space(1))) v4i* as1_v4i_ptr;
typedef __attribute__((address_space(3))) v4i* as3_v4i_ptr;

__device__ __forceinline__ void copy16_g2l(const _Float16* gp, _Float16* lp)
{
#if HAVE_ASYNC_LDS
    __builtin_amdgcn_global_load_async_to_lds_b128(
        (as1_v4i_ptr)(__attribute__((address_space(1))) void*)gp,
        (as3_v4i_ptr)(__attribute__((address_space(3))) void*)lp, 0, 0);
#else
    *(half8*)lp = *(const half8*)gp;
#endif
}

__device__ __forceinline__ void wait_tiles_ready()
{
#if HAVE_ASYNC_LDS
    asm volatile("s_wait_asynccnt 0x0" ::: "memory");
#endif
    __syncthreads();
}

// ---------------------------------------------------------------------------
// Shared double-buffered LDS mainloop.
//   A : [M,K] f16 row-major, BT : [N,K] f16 row-major (K-contiguous)
//   Block = 256 thr = 8 waves; block tile 128(M) x NBLK(N); K step 32.
//   Wave tile: 32(M) x NI*16(N)  (waves arranged 4 in M x 2 in N)
// ---------------------------------------------------------------------------
template<int NI, int NBLK>
__device__ __forceinline__ void gemm_mainloop(const _Float16* __restrict__ A,
                                              const _Float16* __restrict__ BT,
                                              int K, int mblk0, int nblk0,
                                              _Float16* lds, v8f (&acc)[2][NI],
                                              int tid)
{
    constexpr int MBLK  = 128;
    constexpr int ATILE = MBLK * LDS_STRIDE;
    constexpr int BTILE = NBLK * LDS_STRIDE;
    constexpr int BUFSZ = ATILE + BTILE;
    constexpr int ASEGS = (MBLK * 4) / 256;          // full iterations, no tail
    constexpr int BSEGS = (NBLK * 4) / 256;
    static_assert((MBLK * 4) % 256 == 0 && (NBLK * 4) % 256 == 0,
                  "copy loops must have uniform trip counts");

    const int wave = tid >> 5;
    const int lane = tid & 31;
    const int l15  = lane & 15;
    const int hh   = lane >> 4;
    const int mwave = (wave >> 1) * 32;
    const int nwave = (wave & 1) * (NI * 16);

    // Uniform, divergence-free tile staging: every thread copies a fixed set
    // of 16B segments (row = s>>2, seg = s&3).
    auto load_tiles = [&](int buf, int kc) {
        _Float16* la = lds + buf * BUFSZ;
        _Float16* lb = la + ATILE;
#pragma unroll
        for (int i = 0; i < ASEGS; ++i) {
            const int s = i * 256 + tid;
            const int row = s >> 2, seg = s & 3;
            copy16_g2l(A + (size_t)(mblk0 + row) * K + kc + seg * 8,
                       la + row * LDS_STRIDE + seg * 8);
        }
#pragma unroll
        for (int i = 0; i < BSEGS; ++i) {
            const int s = i * 256 + tid;
            const int row = s >> 2, seg = s & 3;
            copy16_g2l(BT + (size_t)(nblk0 + row) * K + kc + seg * 8,
                       lb + row * LDS_STRIDE + seg * 8);
        }
    };

    load_tiles(0, 0);
    int buf = 0;
    for (int kc = 0; kc < K; kc += 32) {
        wait_tiles_ready();                  // tile(buf) visible to all waves
        if (kc + 32 < K) load_tiles(buf ^ 1, kc + 32);   // prefetch next tile

        const _Float16* la = lds + buf * BUFSZ;
        const _Float16* lb = la + ATILE;

        Frag16 fa[2], fb[NI];
#pragma unroll
        for (int mi = 0; mi < 2; ++mi) {
            const _Float16* p = la + (mwave + mi * 16 + l15) * LDS_STRIDE + 8 * hh;
            fa[mi].h[0] = *(const half8*)(p);        // K = kc+8h .. +7
            fa[mi].h[1] = *(const half8*)(p + 16);   // K = kc+16+8h .. +7
        }
#pragma unroll
        for (int ni = 0; ni < NI; ++ni) {
            const _Float16* p = lb + (nwave + ni * 16 + l15) * LDS_STRIDE + 8 * hh;
            fb[ni].h[0] = *(const half8*)(p);        // K = kc+8h .. +7
            fb[ni].h[1] = *(const half8*)(p + 16);   // K = kc+16+8h .. +7
        }
#pragma unroll
        for (int mi = 0; mi < 2; ++mi)
#pragma unroll
            for (int ni = 0; ni < NI; ++ni)
                acc[mi][ni] = __builtin_amdgcn_wmma_f32_16x16x32_f16(
                    false, fa[mi].v, false, fb[ni].v,
                    (short)0, acc[mi][ni], false, false);
        buf ^= 1;
    }
}

// ---------------------------------------------------------------------------
// GEMM1: Hout[m,n] = f16( tanh( Z@W1 + cond ) )   M=N=2048 (block 128x128)
// ---------------------------------------------------------------------------
__global__ __launch_bounds__(256)
void gemm1_tanh_kernel(const _Float16* __restrict__ A,
                       const _Float16* __restrict__ BT,
                       const float* __restrict__ cond,
                       _Float16* __restrict__ Hout,
                       int N, int K)
{
    extern __shared__ _Float16 lds[];
    const int tid  = threadIdx.x;
    const int mblk0 = blockIdx.y * 128;
    const int nblk0 = blockIdx.x * 128;

    v8f acc[2][4] = {};
    gemm_mainloop<4, 128>(A, BT, K, mblk0, nblk0, lds, acc, tid);

    const int wave = tid >> 5, lane = tid & 31;
    const int l15 = lane & 15, hh = lane >> 4;
    const int mwave = (wave >> 1) * 32;
    const int nwave = (wave & 1) * 64;

#pragma unroll
    for (int mi = 0; mi < 2; ++mi)
#pragma unroll
        for (int ni = 0; ni < 4; ++ni) {
            const int n = nblk0 + nwave + ni * 16 + l15;
#pragma unroll
            for (int r = 0; r < 8; ++r) {
                const int m = mblk0 + mwave + mi * 16 + 8 * hh + r;
                float v = acc[mi][ni][r] + cond[(size_t)m * N + n];
                Hout[(size_t)m * N + n] = (_Float16)tanhf(v);
            }
        }
}

// ---------------------------------------------------------------------------
// GEMM2: Kout[m,n] = H@W2 + b2   (f32 out; N=512, block 128x64)
// ---------------------------------------------------------------------------
__global__ __launch_bounds__(256)
void gemm2_bias_kernel(const _Float16* __restrict__ A,
                       const _Float16* __restrict__ BT,
                       const float* __restrict__ b2,
                       float* __restrict__ Kout,
                       int N, int K)
{
    extern __shared__ _Float16 lds[];
    const int tid  = threadIdx.x;
    const int mblk0 = blockIdx.y * 128;
    const int nblk0 = blockIdx.x * 64;

    v8f acc[2][2] = {};
    gemm_mainloop<2, 64>(A, BT, K, mblk0, nblk0, lds, acc, tid);

    const int wave = tid >> 5, lane = tid & 31;
    const int l15 = lane & 15, hh = lane >> 4;
    const int mwave = (wave >> 1) * 32;
    const int nwave = (wave & 1) * 32;

#pragma unroll
    for (int mi = 0; mi < 2; ++mi)
#pragma unroll
        for (int ni = 0; ni < 2; ++ni) {
            const int n = nblk0 + nwave + ni * 16 + l15;
            const float bias = b2[n];
#pragma unroll
            for (int r = 0; r < 8; ++r) {
                const int m = mblk0 + mwave + mi * 16 + 8 * hh + r;
                Kout[(size_t)m * N + n] = acc[mi][ni][r] + bias;
            }
        }
}

// ---------------------------------------------------------------------------
// cond[b,h] = b1[h] + sum_k s[b,k] * Ws[k,h]      (one-time, tiny GEMM)
// ---------------------------------------------------------------------------
__global__ void cond_kernel(const float* __restrict__ s,
                            const float* __restrict__ Ws,
                            const float* __restrict__ b1,
                            float* __restrict__ cond)
{
    const int idx = blockIdx.x * blockDim.x + threadIdx.x;
    if (idx >= BDIM * HDIM) return;
    const int b = idx / HDIM;
    const int h = idx % HDIM;
    float a = b1[h];
    const float* sr = s + (size_t)b * SDIM;
#pragma unroll 8
    for (int k = 0; k < SDIM; ++k)
        a += sr[k] * Ws[(size_t)k * HDIM + h];
    cond[idx] = a;
}

// ---------------------------------------------------------------------------
// dst[c,r] = f16(src[r,c])  — transpose + convert weights (one-time)
// ---------------------------------------------------------------------------
__global__ void transpose_f32_to_f16(const float* __restrict__ src,
                                     _Float16* __restrict__ dst,
                                     int R, int C)
{
    const int idx = blockIdx.x * blockDim.x + threadIdx.x;
    if (idx >= R * C) return;
    const int r = idx / C;
    const int c = idx % C;
    dst[(size_t)c * R + r] = (_Float16)src[idx];
}

// ---------------------------------------------------------------------------
// out[i] = f16( z[i] + coefmul*(t[i0+1]-t[i0]) * k[i] )   (stage input)
// ---------------------------------------------------------------------------
__global__ void axpy_to_f16(const float* __restrict__ z,
                            const float* __restrict__ k,
                            _Float16* __restrict__ out,
                            const float* __restrict__ tarr,
                            int i0, float coefmul, int n)
{
    const int idx = blockIdx.x * blockDim.x + threadIdx.x;
    if (idx >= n) return;
    const float c = coefmul * (tarr[i0 + 1] - tarr[i0]);
    out[idx] = (_Float16)(z[idx] + c * k[idx]);
}

// ---------------------------------------------------------------------------
// z += (t[i0+1]-t[i0]) * coef * (k1 + 2k2 + 2k3 + k4)     (RK4 combine)
// ---------------------------------------------------------------------------
__global__ void rk4_combine(float* __restrict__ z,
                            const float* __restrict__ k1,
                            const float* __restrict__ k2,
                            const float* __restrict__ k3,
                            const float* __restrict__ k4,
                            const float* __restrict__ tarr,
                            int i0, float coef, int n)
{
    const int idx = blockIdx.x * blockDim.x + threadIdx.x;
    if (idx >= n) return;
    const float dt6 = coef * (tarr[i0 + 1] - tarr[i0]);
    z[idx] += dt6 * (k1[idx] + 2.0f * k2[idx] + 2.0f * k3[idx] + k4[idx]);
}

// ---------------------------------------------------------------------------
// Host-side orchestration (graph-capture safe: fixed launch sequence)
// ---------------------------------------------------------------------------
extern "C" void kernel_launch(void* const* d_in, const int* in_sizes, int n_in,
                              void* d_out, int out_size, void* d_ws, size_t ws_size,
                              hipStream_t stream)
{
    const float* z0 = (const float*)d_in[0];   // [B,D]
    const float* s  = (const float*)d_in[1];   // [B,S]
    const float* t  = (const float*)d_in[2];   // [T]
    const float* W1 = (const float*)d_in[3];   // [D,H]
    const float* Ws = (const float*)d_in[4];   // [S,H]
    const float* b1 = (const float*)d_in[5];   // [H]
    const float* W2 = (const float*)d_in[6];   // [H,D]
    const float* b2 = (const float*)d_in[7];   // [D]
    float* out = (float*)d_out;                // [T,B,D]

    // Workspace carve-up (256-byte aligned slabs)
    char* ws = (char*)d_ws;
    size_t off = 0;
    auto carve = [&](size_t bytes) {
        void* p = ws + off;
        off += (bytes + 255) & ~(size_t)255;
        return p;
    };
    float*    cond = (float*)   carve((size_t)BDIM * HDIM * 4);  // 16 MB
    _Float16* W1T  = (_Float16*)carve((size_t)HDIM * DDIM * 2);  //  2 MB
    _Float16* W2T  = (_Float16*)carve((size_t)DDIM * HDIM * 2);  //  2 MB
    _Float16* zs   = (_Float16*)carve((size_t)BDIM * DDIM * 2);  //  2 MB
    _Float16* Hb   = (_Float16*)carve((size_t)BDIM * HDIM * 2);  //  8 MB
    float*    z    = (float*)   carve((size_t)BDIM * DDIM * 4);  //  4 MB
    float*    k1   = (float*)   carve((size_t)BDIM * DDIM * 4);
    float*    k2   = (float*)   carve((size_t)BDIM * DDIM * 4);
    float*    k3   = (float*)   carve((size_t)BDIM * DDIM * 4);
    float*    k4   = (float*)   carve((size_t)BDIM * DDIM * 4);
    (void)ws_size; (void)in_sizes; (void)n_in; (void)out_size;

    const int NZ = BDIM * DDIM;                      // 1,048,576
    const int EW_BLOCKS = (NZ + 255) / 256;

    // --- one-time preprocessing --------------------------------------------
    cond_kernel<<<(BDIM * HDIM + 255) / 256, 256, 0, stream>>>(s, Ws, b1, cond);
    transpose_f32_to_f16<<<(DDIM * HDIM + 255) / 256, 256, 0, stream>>>(W1, W1T, DDIM, HDIM);
    transpose_f32_to_f16<<<(HDIM * DDIM + 255) / 256, 256, 0, stream>>>(W2, W2T, HDIM, DDIM);

    // z = z0 ; out[0] = z0
    (void)hipMemcpyAsync(z,   z0, (size_t)NZ * 4, hipMemcpyDeviceToDevice, stream);
    (void)hipMemcpyAsync(out, z0, (size_t)NZ * 4, hipMemcpyDeviceToDevice, stream);

    // GEMM grids and dynamic LDS sizes (double-buffered A+B tiles)
    dim3 g1(HDIM / 128, BDIM / 128);   // (16,16)
    dim3 g2(DDIM / 64,  BDIM / 128);   // (8,16)
    const size_t shm1 = 2 * (size_t)(128 + 128) * LDS_STRIDE * sizeof(_Float16); // 40960
    const size_t shm2 = 2 * (size_t)(128 + 64)  * LDS_STRIDE * sizeof(_Float16); // 30720

    const float inv_steps = 1.0f / (float)STEPS_PER_INTERVAL;

    auto eval_f = [&](const _Float16* zin, float* kout) {
        gemm1_tanh_kernel<<<g1, 256, shm1, stream>>>(zin, W1T, cond, Hb, HDIM, DDIM);
        gemm2_bias_kernel<<<g2, 256, shm2, stream>>>(Hb, W2T, b2, kout, DDIM, HDIM);
    };

    for (int iv = 0; iv < TDIM - 1; ++iv) {
        for (int st = 0; st < STEPS_PER_INTERVAL; ++st) {
            // stage 1: k1 = f(z)
            axpy_to_f16<<<EW_BLOCKS, 256, 0, stream>>>(z, z, zs, t, iv, 0.0f, NZ);
            eval_f(zs, k1);
            // stage 2: k2 = f(z + dt/2 * k1)
            axpy_to_f16<<<EW_BLOCKS, 256, 0, stream>>>(z, k1, zs, t, iv, 0.5f * inv_steps, NZ);
            eval_f(zs, k2);
            // stage 3: k3 = f(z + dt/2 * k2)
            axpy_to_f16<<<EW_BLOCKS, 256, 0, stream>>>(z, k2, zs, t, iv, 0.5f * inv_steps, NZ);
            eval_f(zs, k3);
            // stage 4: k4 = f(z + dt * k3)
            axpy_to_f16<<<EW_BLOCKS, 256, 0, stream>>>(z, k3, zs, t, iv, 1.0f * inv_steps, NZ);
            eval_f(zs, k4);
            // z += dt/6 * (k1 + 2k2 + 2k3 + k4)
            rk4_combine<<<EW_BLOCKS, 256, 0, stream>>>(z, k1, k2, k3, k4, t, iv,
                                                       inv_steps / 6.0f, NZ);
        }
        (void)hipMemcpyAsync(out + (size_t)(iv + 1) * NZ, z, (size_t)NZ * 4,
                             hipMemcpyDeviceToDevice, stream);
    }
}
